// ROIHeadVgg16_15504831939155
// MI455X (gfx1250) — matrix operand order repaired and checked by reading, not compiled, not run on hardware
//
#include <hip/hip_runtime.h>

// ---------------------------------------------------------------------------
// ROI head (Faster R-CNN VGG16) for MI455X / gfx1250, wave32 + WMMA bf16.
//
//   1) roi_pool_kernel : [1,512,38,50] fp32 + 300 ROIs -> pooled bf16 [320][25088]
//   2) gemm_bf16_kernel: fc6 = relu(pooled @ W1 + b1)  -> bf16 [320][4096]
//   3) gemm_bf16_kernel: fc7 = relu(fc6   @ W2 + b2)  -> bf16 [320][4096]
//   4) heads_kernel    : deltas = fc7 @ Wp + bp ; scores = fc7 @ Ws + bs
//
// GEMMs stream fp32 weights from HBM once (411MB W1 -> ~18us @ 23.3TB/s),
// convert fp32->bf16 in-register, stage transposed in LDS, and run
// v_wmma_f32_16x16x32_bf16. The A (activation) tile is staged into LDS with
// CDNA5 async copies (global_load_async_to_lds_b128 / s_wait_asynccnt) so
// each block fetches it once per K-step and the steady-state loop has no
// 64-bit address math and no per-WMMA global-load stalls.
// ---------------------------------------------------------------------------

typedef __bf16 v16bf __attribute__((ext_vector_type(16)));
typedef float  v8f   __attribute__((ext_vector_type(8)));

union FragU {
    uint4 q[2];   // 32 bytes = 8 VGPRs
    v16bf v;      // 16 bf16 elements
};

__device__ __forceinline__ unsigned short f2bf(float f) {
    // round-to-nearest-even fp32 -> bf16
    unsigned u = __float_as_uint(f);
    u += 0x7FFFu + ((u >> 16) & 1u);
    return (unsigned short)(u >> 16);
}

__device__ __forceinline__ unsigned pack2(float lo, float hi) {
    // bf16 pair: lo -> bits[15:0] (even K), hi -> bits[31:16] (odd K)
    return (unsigned)f2bf(lo) | ((unsigned)f2bf(hi) << 16);
}

// ---------------------------------------------------------------------------
// Kernel 1: Caffe-style ROI max pooling -> bf16, row-major [320][25088]
// ---------------------------------------------------------------------------
#define POOL_C   512
#define POOL_H   38
#define POOL_W   50
#define POOL_D   25088    // 512*7*7
#define M_ROWS   320      // 300 ROIs padded to 20 WMMA m-tiles
#define N_ROIS   300

__global__ __launch_bounds__(256) void roi_pool_kernel(
        const float* __restrict__ x, const float* __restrict__ rois,
        const int* __restrict__ ridx, unsigned short* __restrict__ pooled)
{
    const int r   = blockIdx.x;
    const int tid = threadIdx.x;
    unsigned short* dst = pooled + (size_t)r * POOL_D;

    if (r >= N_ROIS) {               // zero-pad rows 300..319
        for (int o = tid; o < POOL_D; o += 256) dst[o] = 0;
        return;
    }

    const float y1 = rois[r * 4 + 0];
    const float x1 = rois[r * 4 + 1];
    const float y2 = rois[r * 4 + 2];
    const float x2 = rois[r * 4 + 3];
    const int sw = (int)roundf(x1 * 0.0625f);
    const int sh = (int)roundf(y1 * 0.0625f);
    const int ew = (int)roundf(x2 * 0.0625f);
    const int eh = (int)roundf(y2 * 0.0625f);
    const float bh = fmaxf((float)(eh - sh + 1), 1.0f) * (1.0f / 7.0f);
    const float bw = fmaxf((float)(ew - sw + 1), 1.0f) * (1.0f / 7.0f);

    const float* fm = x + (size_t)ridx[r] * (POOL_C * POOL_H * POOL_W);

    int hs[7], he[7], wss[7], wee[7];
#pragma unroll
    for (int p = 0; p < 7; ++p) {
        int a = (int)floorf((float)p * bh) + sh;
        int b = (int)ceilf((float)(p + 1) * bh) + sh;
        hs[p] = min(max(a, 0), POOL_H);
        he[p] = min(max(b, 0), POOL_H);
        a = (int)floorf((float)p * bw) + sw;
        b = (int)ceilf((float)(p + 1) * bw) + sw;
        wss[p] = min(max(a, 0), POOL_W);
        wee[p] = min(max(b, 0), POOL_W);
    }

    for (int o = tid; o < POOL_D; o += 256) {
        const int c   = o / 49;
        const int rem = o - c * 49;
        const int ph  = rem / 7;
        const int pw  = rem - ph * 7;
        const int h0 = hs[ph], h1 = he[ph];
        const int w0 = wss[pw], w1 = wee[pw];
        float m = 0.0f;
        if (h0 < h1 && w0 < w1) {
            m = -3.402823e38f;
            const float* base = fm + (size_t)c * (POOL_H * POOL_W);
            for (int h = h0; h < h1; ++h) {
                const float* rowp = base + h * POOL_W;
                for (int w = w0; w < w1; ++w) m = fmaxf(m, rowp[w]);
            }
        }
        dst[o] = f2bf(m);
    }
}

// ---------------------------------------------------------------------------
// Kernel 2: bf16 WMMA GEMM + bias + ReLU -> bf16
//   out[320][4096] = relu(A[320][K] (bf16) @ Wt[K][4096] (fp32) + bias)
//   Grid: 128 blocks (one per 32-col slice of N -> W streamed from HBM once).
//   Block: 256 threads = 8 waves: 4 m-waves (5 WMMA tiles) x 2 n-waves.
//   Per K-step (BK=32), double-buffered:
//     - W tile: coalesced fp32 loads, packed to bf16 pairs, stored transposed
//       in LDS (stride 20 dwords -> conflict-free ds_load_b128 fragments)
//     - A tile: 320x32 bf16 staged by async copies global_load_async_to_lds_b128
//       (ASYNCcnt), row stride padded to 80B -> conflict-free fragment reads
//     - 5 x v_wmma_f32_16x16x32_bf16 per wave, fragments batched in registers
// ---------------------------------------------------------------------------
#define N_TOT 4096
#define BN    32
#define BK    32
#define BS_STRIDE   20   // dwords per B column: 16 used + 4 pad
#define LDSA_STRIDE 80   // bytes per A row: 64 used + 16 pad

__global__ __launch_bounds__(256) void gemm_bf16_kernel(
        const unsigned short* __restrict__ A, const float* __restrict__ Wt,
        const float* __restrict__ bias, unsigned short* __restrict__ out,
        const int K)
{
    __shared__ __align__(16) unsigned char ldsA[2][M_ROWS * LDSA_STRIDE]; // 51200B
    __shared__ __align__(16) unsigned int  ldsB[2][BN * BS_STRIDE];       //  5120B

    const int tid   = threadIdx.x;
    const int lane  = tid & 31;
    const int l15   = lane & 15;
    const int lhi   = lane >> 4;        // 0 or 1
    const int wid   = tid >> 5;         // 0..7
    const int wn    = wid & 1;          // n-wave: 2 x 16 cols
    const int wm    = wid >> 1;         // m-wave: 4 x 80 rows
    const int nBase = blockIdx.x * BN;
    const int mBase = wm * 80;

    // W-tile loader assignment: 16 col-pairs x 16 k-pairs
    const int wn2 = (tid & 15) * 2;     // column pair base 0..30
    const int wkk = (tid >> 4) * 2;     // k pair base     0..30

    // A async-copy descriptors: 1280 chunks of 16B per K-step, 5 per thread
    unsigned    aLds[5];
    const char* aGlb[5];
#pragma unroll
    for (int i = 0; i < 5; ++i) {
        const int c    = tid + i * 256;
        const int row  = c >> 2;        // 0..319
        const int part = c & 3;         // 16B sub-chunk within 64B row slice
        aLds[i] = (unsigned)(uintptr_t)(&ldsA[0][row * LDSA_STRIDE + part * 16]);
        aGlb[i] = (const char*)(A + (size_t)row * K) + part * 16;
    }
    const unsigned aBufDelta = (unsigned)(M_ROWS * LDSA_STRIDE);

    v8f acc[5];
#pragma unroll
    for (int t = 0; t < 5; ++t)
#pragma unroll
        for (int v = 0; v < 8; ++v) acc[t][v] = 0.0f;

    const int nSteps = K / BK;

    // ---- prologue: stage K-step 0 into buffer 0 ----
    {
        const float* w0 = Wt + (size_t)wkk * N_TOT + nBase + wn2;
        float2 r0 = *(const float2*)w0;
        float2 r1 = *(const float2*)(w0 + N_TOT);
#pragma unroll
        for (int i = 0; i < 5; ++i) {
            asm volatile("global_load_async_to_lds_b128 %0, %1, off"
                         :: "v"(aLds[i]), "v"(aGlb[i]) : "memory");
        }
        ldsB[0][(wn2    ) * BS_STRIDE + (wkk >> 1)] = pack2(r0.x, r1.x);
        ldsB[0][(wn2 + 1) * BS_STRIDE + (wkk >> 1)] = pack2(r0.y, r1.y);
    }
    asm volatile("s_wait_asynccnt 0x0" ::: "memory");
    __syncthreads();

    // ---- steady state: branch-free, last step peeled ----
    for (int s = 0; s < nSteps - 1; ++s) {
        const int buf = s & 1;
        const int nb  = buf ^ 1;

        // issue global loads for next W tile early
        const float* w0 = Wt + ((size_t)(s + 1) * BK + wkk) * N_TOT + nBase + wn2;
        float2 r0 = *(const float2*)w0;
        float2 r1 = *(const float2*)(w0 + N_TOT);
        __builtin_prefetch(w0 + (size_t)BK * N_TOT, 0, 1);

        // issue async copies for next A tile (fire-and-forget, ASYNCcnt)
#pragma unroll
        for (int i = 0; i < 5; ++i) {
            const char*    g = aGlb[i] + (size_t)(s + 1) * (BK * 2);
            const unsigned l = aLds[i] + (nb ? aBufDelta : 0u);
            asm volatile("global_load_async_to_lds_b128 %0, %1, off"
                         :: "v"(l), "v"(g) : "memory");
        }

        // B fragment from LDS
        FragU bfrag;
        const uint4* bp_ =
            (const uint4*)&ldsB[buf][(wn * 16 + l15) * BS_STRIDE + lhi * 4];
        bfrag.q[0] = bp_[0];
        bfrag.q[1] = bp_[2];

        // batch all 5 A fragments into registers, then run 5 WMMAs
        FragU af[5];
#pragma unroll
        for (int t = 0; t < 5; ++t) {
            const uint4* ap = (const uint4*)
                &ldsA[buf][(mBase + t * 16 + l15) * LDSA_STRIDE + lhi * 16];
            af[t].q[0] = ap[0];
            af[t].q[1] = ap[2];   // +32 bytes
        }
#pragma unroll
        for (int t = 0; t < 5; ++t) {
            acc[t] = __builtin_amdgcn_wmma_f32_16x16x32_bf16(
                false, af[t].v, false, bfrag.v, (short)0, acc[t], false, false);
        }

        // stage next B tile (bf16 transposed) into the other buffer
        ldsB[nb][(wn2    ) * BS_STRIDE + (wkk >> 1)] = pack2(r0.x, r1.x);
        ldsB[nb][(wn2 + 1) * BS_STRIDE + (wkk >> 1)] = pack2(r0.y, r1.y);

        asm volatile("s_wait_asynccnt 0x0" ::: "memory");
        __syncthreads();
    }

    // ---- final K-step: compute only ----
    {
        const int buf = (nSteps - 1) & 1;
        FragU bfrag;
        const uint4* bp_ =
            (const uint4*)&ldsB[buf][(wn * 16 + l15) * BS_STRIDE + lhi * 4];
        bfrag.q[0] = bp_[0];
        bfrag.q[1] = bp_[2];
        FragU af[5];
#pragma unroll
        for (int t = 0; t < 5; ++t) {
            const uint4* ap = (const uint4*)
                &ldsA[buf][(mBase + t * 16 + l15) * LDSA_STRIDE + lhi * 16];
            af[t].q[0] = ap[0];
            af[t].q[1] = ap[2];
        }
#pragma unroll
        for (int t = 0; t < 5; ++t) {
            acc[t] = __builtin_amdgcn_wmma_f32_16x16x32_bf16(
                false, af[t].v, false, bfrag.v, (short)0, acc[t], false, false);
        }
    }

    // ---- epilogue: bias + relu + bf16 store ----
    const int gcol = nBase + wn * 16 + l15;
    const float bv = bias[gcol];
#pragma unroll
    for (int t = 0; t < 5; ++t) {
        const int rb = mBase + t * 16 + lhi * 8;   // C layout: VGPR v -> row rb+v
#pragma unroll
        for (int v = 0; v < 8; ++v) {
            float val = acc[t][v] + bv;
            val = val > 0.0f ? val : 0.0f;
            out[(size_t)(rb + v) * N_TOT + gcol] = f2bf(val);
        }
    }
}

// ---------------------------------------------------------------------------
// Kernel 3: head projections. One block per ROI; fc7 row cached in LDS.
// Thread n < 84 -> delta column n (Wp), 84 <= n < 105 -> score column n-84 (Ws).
// ---------------------------------------------------------------------------
#define N_DELTA 84
#define N_SCORE 21

__global__ __launch_bounds__(128) void heads_kernel(
        const unsigned short* __restrict__ fc7,
        const float* __restrict__ Wsc, const float* __restrict__ bsc,
        const float* __restrict__ Wpd, const float* __restrict__ bpd,
        float* __restrict__ outp)
{
    __shared__ float xs[N_TOT];
    const int r   = blockIdx.x;
    const int tid = threadIdx.x;

    const unsigned short* row = fc7 + (size_t)r * N_TOT;
    for (int k = tid; k < N_TOT; k += 128)
        xs[k] = __uint_as_float(((unsigned)row[k]) << 16);
    __syncthreads();

    const int n = tid;
    if (n >= N_DELTA + N_SCORE) return;

    const float* Wc;
    float acc;
    int ldn;
    if (n < N_DELTA) { Wc = Wpd + n;             acc = bpd[n];           ldn = N_DELTA; }
    else             { Wc = Wsc + (n - N_DELTA); acc = bsc[n - N_DELTA]; ldn = N_SCORE; }

#pragma unroll 8
    for (int k = 0; k < N_TOT; ++k)
        acc += xs[k] * Wc[(size_t)k * ldn];

    if (n < N_DELTA) outp[(size_t)r * N_DELTA + n] = acc;
    else             outp[(size_t)N_ROIS * N_DELTA + (size_t)r * N_SCORE + (n - N_DELTA)] = acc;
}

// ---------------------------------------------------------------------------
// Launcher
// ---------------------------------------------------------------------------
extern "C" void kernel_launch(void* const* d_in, const int* in_sizes, int n_in,
                              void* d_out, int out_size, void* d_ws, size_t ws_size,
                              hipStream_t stream)
{
    (void)in_sizes; (void)n_in; (void)out_size; (void)ws_size;

    const float* x    = (const float*)d_in[0];
    const float* rois = (const float*)d_in[1];
    const int*   ridx = (const int*)  d_in[2];
    const float* W1   = (const float*)d_in[3];
    const float* b1   = (const float*)d_in[4];
    const float* W2   = (const float*)d_in[5];
    const float* b2   = (const float*)d_in[6];
    const float* Wsc  = (const float*)d_in[7];
    const float* bsc  = (const float*)d_in[8];
    const float* Wpd  = (const float*)d_in[9];
    const float* bpd  = (const float*)d_in[10];
    float* outp = (float*)d_out;

    char* ws = (char*)d_ws;
    const size_t pooled_bytes = (size_t)M_ROWS * POOL_D * 2;   // 16,056,320
    const size_t fc_bytes     = (size_t)M_ROWS * N_TOT * 2;    //  2,621,440
    unsigned short* pooled = (unsigned short*)ws;
    unsigned short* fc6    = (unsigned short*)(ws + pooled_bytes);
    unsigned short* fc7    = (unsigned short*)(ws + pooled_bytes + fc_bytes);

    hipLaunchKernelGGL(roi_pool_kernel, dim3(M_ROWS), dim3(256), 0, stream,
                       x, rois, ridx, pooled);
    hipLaunchKernelGGL(gemm_bf16_kernel, dim3(N_TOT / BN), dim3(256), 0, stream,
                       pooled, W1, b1, fc6, POOL_D);
    hipLaunchKernelGGL(gemm_bf16_kernel, dim3(N_TOT / BN), dim3(256), 0, stream,
                       fc6, W2, b2, fc7, N_TOT);
    hipLaunchKernelGGL(heads_kernel, dim3(N_ROIS), dim3(128), 0, stream,
                       fc7, Wsc, bsc, Wpd, bpd, outp);
}
